// MambaBlock_82875688943957
// MI455X (gfx1250) — matrix operand order repaired and checked by reading, not compiled, not run on hardware
//
#include <hip/hip_runtime.h>
#include <hip/hip_bf16.h>
#include <math.h>

// ---------------------------------------------------------------------------
// Mamba block for MI455X (gfx1250, wave32).
// GEMMs: v_wmma_f32_16x16x32_bf16, 32x32 per-wave tiles (4 WMMA chains, each
// fragment reused twice), padded B^T so the K-loop has zero bounds checks.
// Scan: register-resident h, LDS-staged B/C, shfl pair-reduction, fused gate.
// ---------------------------------------------------------------------------

typedef __bf16 bf16_t;
typedef bf16_t v16bf __attribute__((ext_vector_type(16)));
typedef bf16_t v8bf  __attribute__((ext_vector_type(8)));
typedef float  v8f   __attribute__((ext_vector_type(8)));

#define DEV static __device__ __forceinline__

// Problem constants (fixed by reference setup_inputs)
#define CB   4
#define CL   2048
#define CDIM 1024
#define CDI  2048            // d_inner
#define CDS  16              // d_state
#define CDR  64              // dt_rank
#define CM   (CB * CL)       // 8192 rows
#define CXZ  (2 * CDI)       // 4096
#define CDBC (CDR + 2 * CDS) // 96
#define CDBCP 128            // dbc N padded to block-tile

// ---------------------------------------------------------------------------
// Helpers
// ---------------------------------------------------------------------------
DEV float sigmoidf_fast(float v) { return 1.f / (1.f + __expf(-v)); }

// A fragment: 16x32 bf16 from row-major A, row = base + (lane&15).
// lanes 0-15 hold K = k..k+7 (e0..7) and k+16..k+23 (e8..15);
// lanes 16-31 hold K = k+8..k+15 and k+24..k+31.  (ISA 7.12.2)
DEV v16bf load_a_frag(const bf16_t* __restrict__ rowp, int k, int khalf) {
  v16bf r;
  const bf16_t* p = rowp + k + khalf * 8;
  *(v8bf*)&r       = *(const v8bf*)p;
  *((v8bf*)&r + 1) = *(const v8bf*)(p + 16);
  return r;
}

// B fragment from padded B^T [NP,K]: column n = row of Bt; lanes 0-15 hold
// K=k..k+7 and k+16..k+23, lanes 16-31 hold K=k+8..k+15 and k+24..k+31.
DEV v16bf load_b_frag(const bf16_t* __restrict__ colp, int k, int khalf) {
  v16bf r;
  const bf16_t* p = colp + k + khalf * 8;
  *(v8bf*)&r       = *(const v8bf*)p;
  *((v8bf*)&r + 1) = *(const v8bf*)(p + 16);
  return r;
}

template <int EPI>
DEV float epilogue(float acc, const float* __restrict__ aux, size_t idx, int col) {
  if (EPI == 1) {                       // delta = softplus(acc + dt_b)
    float v = acc + aux[col];
    return (v > 20.f) ? v : log1pf(__expf(v));
  }
  if (EPI == 2) return aux[idx] + acc;  // residual add (out = x + y@Wo)
  return acc;
}

DEV v8f wmma_bf16(v16bf a, v16bf b, v8f c) {
  return __builtin_amdgcn_wmma_f32_16x16x32_bf16(false, a, false, b,
                                                 (short)0, c, false, false);
}

// ---------------------------------------------------------------------------
// GEMM: C[M,N] = A[M,K](bf16) @ Bt[NP,K](bf16)^T, fp32 accumulate via WMMA.
// 256 threads = 8 waves arranged 2(M) x 4(N); each wave owns a 32x32 tile
// (4 accumulators, every A/B fragment feeds two WMMAs). Block tile 64x128.
// Bt is padded to NP cols so the K-loop is branch-free; stores guard col<N.
// ---------------------------------------------------------------------------
template <int EPI>
__global__ __launch_bounds__(256)
void gemm_bf16_wmma(const bf16_t* __restrict__ A, const bf16_t* __restrict__ Bt,
                    float* __restrict__ C, int M, int N, int K,
                    const float* __restrict__ aux) {
  const int lane  = threadIdx.x & 31;
  const int wave  = threadIdx.x >> 5;
  const int wm    = wave >> 2;          // 0..1
  const int wn    = wave & 3;           // 0..3
  const int m0    = blockIdx.y * 64 + wm * 32;
  const int n0    = blockIdx.x * 128 + wn * 32;
  const int khalf = lane >> 4;
  const int l16   = lane & 15;

  const bf16_t* pa0 = A  + (size_t)(m0 + l16)      * K;
  const bf16_t* pa1 = A  + (size_t)(m0 + 16 + l16) * K;
  const bf16_t* pb0 = Bt + (size_t)(n0 + l16)      * K;
  const bf16_t* pb1 = Bt + (size_t)(n0 + 16 + l16) * K;

  v8f acc00 = {}, acc01 = {}, acc10 = {}, acc11 = {};
  for (int k = 0; k < K; k += 32) {
    __builtin_prefetch(pa0 + k + 64, 0, 0);
    __builtin_prefetch(pb0 + k + 64, 0, 0);
    v16bf a0 = load_a_frag(pa0, k, khalf);
    v16bf a1 = load_a_frag(pa1, k, khalf);
    v16bf b0 = load_b_frag(pb0, k, khalf);
    v16bf b1 = load_b_frag(pb1, k, khalf);
    acc00 = wmma_bf16(a0, b0, acc00);
    acc01 = wmma_bf16(a0, b1, acc01);
    acc10 = wmma_bf16(a1, b0, acc10);
    acc11 = wmma_bf16(a1, b1, acc11);
  }

  const int c0 = n0 + l16, c1 = n0 + 16 + l16;
#pragma unroll
  for (int r = 0; r < 8; r++) {
    int row = m0 + khalf * 8 + r;
    size_t b0i = (size_t)row * N;
    size_t b1i = (size_t)(row + 16) * N;
    if (c0 < N) {
      C[b0i + c0] = epilogue<EPI>(acc00[r], aux, b0i + c0, c0);
      C[b1i + c0] = epilogue<EPI>(acc10[r], aux, b1i + c0, c0);
    }
    if (c1 < N) {
      C[b0i + c1] = epilogue<EPI>(acc01[r], aux, b0i + c1, c1);
      C[b1i + c1] = epilogue<EPI>(acc11[r], aux, b1i + c1, c1);
    }
  }
}

// ---------------------------------------------------------------------------
// RMSNorm + bf16 cast: one block per row of x[8192,1024].
// ---------------------------------------------------------------------------
__global__ __launch_bounds__(256)
void rmsnorm_cast(const float* __restrict__ x, const float* __restrict__ w,
                  bf16_t* __restrict__ xn) {
  __shared__ float red[8];
  const int row = blockIdx.x;
  const float* xr = x + (size_t)row * CDIM;
  float ss = 0.f;
  for (int i = threadIdx.x; i < CDIM; i += 256) { float v = xr[i]; ss += v * v; }
#pragma unroll
  for (int off = 16; off > 0; off >>= 1) ss += __shfl_xor(ss, off, 32);
  const int lane = threadIdx.x & 31, wv = threadIdx.x >> 5;
  if (lane == 0) red[wv] = ss;
  __syncthreads();
  if (wv == 0) {
    float t = (lane < 8) ? red[lane] : 0.f;
#pragma unroll
    for (int off = 4; off > 0; off >>= 1) t += __shfl_xor(t, off, 32);
    if (lane == 0) red[0] = t;
  }
  __syncthreads();
  const float inv = rsqrtf(red[0] / (float)CDIM + 1e-5f);
  for (int i = threadIdx.x; i < CDIM; i += 256)
    xn[(size_t)row * CDIM + i] = (bf16_t)(xr[i] * inv * w[i]);
}

// ---------------------------------------------------------------------------
// Weight cast + transpose with zero padding: W[K,N] f32 -> Wt[NP,K] bf16.
// ---------------------------------------------------------------------------
__global__ __launch_bounds__(256)
void cast_transpose(const float* __restrict__ W, bf16_t* __restrict__ Wt,
                    int K, int N, int NP) {
  size_t idx = (size_t)blockIdx.x * blockDim.x + threadIdx.x;
  if (idx >= (size_t)K * NP) return;
  int k = (int)(idx / NP), n = (int)(idx % NP);
  float v = (n < N) ? W[(size_t)k * N + n] : 0.f;
  Wt[(size_t)n * K + k] = (bf16_t)v;
}

// dt slice cast: dbc[:, 0:64] f32 -> dt bf16 [8192,64]
__global__ __launch_bounds__(256)
void cast_dt(const float* __restrict__ dbc, bf16_t* __restrict__ dt) {
  size_t idx = (size_t)blockIdx.x * blockDim.x + threadIdx.x; // 8192*64
  int m = (int)(idx >> 6), j = (int)(idx & 63);
  dt[idx] = (bf16_t)dbc[(size_t)m * CDBC + j];
}

// ---------------------------------------------------------------------------
// Causal depthwise conv (K=4) + SiLU. Input: xi half of xz[8192,4096].
// Writes xi fp32 (for scan) and xi bf16 (for x_proj GEMM).
// ---------------------------------------------------------------------------
__global__ __launch_bounds__(256)
void conv_silu(const float* __restrict__ xz, const float* __restrict__ cw,
               const float* __restrict__ cb, float* __restrict__ xi,
               bf16_t* __restrict__ xi_bf) {
  size_t idx = (size_t)blockIdx.x * blockDim.x + threadIdx.x; // 8192*2048
  int c = (int)(idx % CDI);
  int m = (int)(idx / CDI);
  int l = m % CL;
  float acc = cb[c];
#pragma unroll
  for (int j = 0; j < 4; j++) {
    int li = l - 3 + j;
    if (li >= 0) acc += xz[(size_t)(m - 3 + j) * CXZ + c] * cw[c * 4 + j];
  }
  float s = acc * sigmoidf_fast(acc);
  xi[idx] = s;
  xi_bf[idx] = (bf16_t)s;
}

// ---------------------------------------------------------------------------
// Selective scan. 2 lanes per channel (8 states each), h in registers,
// B/C staged in LDS per 64-step chunk, pair reduction via shfl_xor,
// fused D-skip + SiLU(z) gate + bf16 cast for the output GEMM.
// Grid: CB * (CDI/128) blocks of 256 threads.
// ---------------------------------------------------------------------------
__global__ __launch_bounds__(256)
void selective_scan(const float* __restrict__ delta, const float* __restrict__ xi,
                    const float* __restrict__ dbc, const float* __restrict__ xz,
                    const float* __restrict__ A_log, const float* __restrict__ Dw,
                    bf16_t* __restrict__ y_out) {
  __shared__ float sB[64 * CDS];
  __shared__ float sC[64 * CDS];
  const int tid   = threadIdx.x;
  const int ch    = tid >> 1;     // 0..127 channel within block
  const int shalf = tid & 1;      // which 8 states
  const int bpb   = CDI / 128;    // blocks per batch
  const int b     = blockIdx.x / bpb;
  const int d     = (blockIdx.x % bpb) * 128 + ch;
  const int s0    = shalf * 8;

  float Arow[8], h[8];
#pragma unroll
  for (int s = 0; s < 8; s++) {
    Arow[s] = -__expf(A_log[(size_t)d * CDS + s0 + s]);
    h[s] = 0.f;
  }
  const float Dv = Dw[d];

  for (int lc = 0; lc < CL; lc += 64) {
    for (int i = tid; i < 64 * CDS; i += 256) {
      int step = i >> 4, s = i & 15;
      const float* rrow = dbc + (size_t)(b * CL + lc + step) * CDBC;
      sB[i] = rrow[CDR + s];
      sC[i] = rrow[CDR + CDS + s];
    }
    __syncthreads();
#pragma unroll 4
    for (int j = 0; j < 64; j++) {
      const int l = lc + j;
      const size_t idx = (size_t)(b * CL + l) * CDI + d;
      const float dv = delta[idx];
      const float xv = xi[idx];
      const float dx = dv * xv;
      float y = 0.f;
#pragma unroll
      for (int s = 0; s < 8; s++) {
        float dA = __expf(dv * Arow[s]);
        h[s] = dA * h[s] + dx * sB[j * CDS + s0 + s];
        y += h[s] * sC[j * CDS + s0 + s];
      }
      y += __shfl_xor(y, 1, 32);        // combine the two 8-state halves
      if (shalf == 0) {
        float yy = y + xv * Dv;
        float zv = xz[(size_t)(b * CL + l) * CXZ + CDI + d];
        y_out[idx] = (bf16_t)(yy * (zv * sigmoidf_fast(zv)));
      }
    }
    __syncthreads();
  }
}

// ---------------------------------------------------------------------------
// Host launcher
// ---------------------------------------------------------------------------
extern "C" void kernel_launch(void* const* d_in, const int* in_sizes, int n_in,
                              void* d_out, int out_size, void* d_ws, size_t ws_size,
                              hipStream_t stream) {
  (void)in_sizes; (void)n_in; (void)out_size; (void)ws_size;
  const float* x        = (const float*)d_in[0];
  const float* rms_w    = (const float*)d_in[1];
  const float* in_proj  = (const float*)d_in[2];
  const float* conv_w   = (const float*)d_in[3];
  const float* conv_b   = (const float*)d_in[4];
  const float* x_proj   = (const float*)d_in[5];
  const float* dt_w     = (const float*)d_in[6];
  const float* dt_b     = (const float*)d_in[7];
  const float* A_log    = (const float*)d_in[8];
  const float* Dw       = (const float*)d_in[9];
  const float* out_proj = (const float*)d_in[10];
  float* out = (float*)d_out;

  char* ws = (char*)d_ws;
  size_t off = 0;
  auto alloc = [&](size_t bytes) -> void* {
    void* p = ws + off;
    off += (bytes + 255) & ~(size_t)255;
    return p;
  };

  bf16_t* xn_bf   = (bf16_t*)alloc((size_t)CM * CDIM * 2);
  bf16_t* inpT    = (bf16_t*)alloc((size_t)CXZ * CDIM * 2);    // [4096,1024]
  bf16_t* xprT    = (bf16_t*)alloc((size_t)CDBCP * CDI * 2);   // [128,2048] padded
  bf16_t* dtwT    = (bf16_t*)alloc((size_t)CDI * CDR * 2);     // [2048,64]
  bf16_t* outpT   = (bf16_t*)alloc((size_t)CDIM * CDI * 2);    // [1024,2048]
  float*  xz      = (float*)alloc((size_t)CM * CXZ * 4);
  float*  xi      = (float*)alloc((size_t)CM * CDI * 4);
  bf16_t* xi_bf   = (bf16_t*)alloc((size_t)CM * CDI * 2);
  float*  dbc     = (float*)alloc((size_t)CM * CDBC * 4);
  bf16_t* dt_bf   = (bf16_t*)alloc((size_t)CM * CDR * 2);
  float*  delta   = (float*)alloc((size_t)CM * CDI * 4);
  bf16_t* y_bf    = (bf16_t*)alloc((size_t)CM * CDI * 2);

  // 1) RMSNorm + cast
  rmsnorm_cast<<<CM, 256, 0, stream>>>(x, rms_w, xn_bf);

  // 2) Weight cast + transpose (Bt padded to multiples of the 128-col tile)
  cast_transpose<<<(CDIM * CXZ) / 256,  256, 0, stream>>>(in_proj,  inpT,  CDIM, CXZ,  CXZ);
  cast_transpose<<<(CDI * CDBCP) / 256, 256, 0, stream>>>(x_proj,   xprT,  CDI,  CDBC, CDBCP);
  cast_transpose<<<(CDR * CDI) / 256,   256, 0, stream>>>(dt_w,     dtwT,  CDR,  CDI,  CDI);
  cast_transpose<<<(CDI * CDIM) / 256,  256, 0, stream>>>(out_proj, outpT, CDI,  CDIM, CDIM);

  // 3) xz = xn @ in_proj          [8192,4096]
  gemm_bf16_wmma<0><<<dim3(CXZ / 128, CM / 64), 256, 0, stream>>>(
      xn_bf, inpT, xz, CM, CXZ, CDIM, nullptr);

  // 4) depthwise causal conv + SiLU -> xi (f32 + bf16)
  conv_silu<<<((size_t)CM * CDI) / 256, 256, 0, stream>>>(xz, conv_w, conv_b, xi, xi_bf);

  // 5) dbc = xi @ x_proj          [8192,96] (padded B^T, stores guarded)
  gemm_bf16_wmma<0><<<dim3(CDBCP / 128, CM / 64), 256, 0, stream>>>(
      xi_bf, xprT, dbc, CM, CDBC, CDI, nullptr);

  // 6) dt slice -> bf16
  cast_dt<<<((size_t)CM * CDR) / 256, 256, 0, stream>>>(dbc, dt_bf);

  // 7) delta = softplus(dt @ dt_w + dt_b)   [8192,2048]
  gemm_bf16_wmma<1><<<dim3(CDI / 128, CM / 64), 256, 0, stream>>>(
      dt_bf, dtwT, delta, CM, CDI, CDR, dt_b);

  // 8) selective scan + D-skip + SiLU(z) gate -> y_bf
  selective_scan<<<CB * (CDI / 128), 256, 0, stream>>>(
      delta, xi, dbc, xz, A_log, Dw, y_bf);

  // 9) out = x + y @ out_proj     [8192,1024]
  gemm_bf16_wmma<2><<<dim3(CDIM / 128, CM / 64), 256, 0, stream>>>(
      y_bf, outpT, out, CM, CDIM, CDI, x);
}
